// CausalMultiHeadedAttention_10385230922562
// MI455X (gfx1250) — matrix operand order repaired and checked by reading, not compiled, not run on hardware
//
#include <hip/hip_runtime.h>

// ---------------------------------------------------------------------------
// MI455X (gfx1250) fused multi-head attention, bf16 WMMA (16x16x32), f32 acc.
//   S=4096, D=768, nh=12, hd=64.  Compute-bound (AI ~1400 FLOP/B) -> all
//   matrix math on v_wmma_f32_16x16x32_bf16; conversions hoisted out of loops.
// Pipeline:
//   k0: one-shot f32->bf16 conversion of x / w_qkv / w_proj (b128 in/out).
//   k1: qkv = x @ w_qkv^T, 32x64 wave tiles (8 wmma per K-chunk, 1.5 b128
//       loads/wmma); scatter q(scaled)/k as [nh][S][hd], v transposed
//       [nh][hd][S] so PV B-fragments are contiguous.
//   k2: fused attention, one wave per (head, 32 query rows); 2-pass softmax
//       (recompute QK^T instead of rescaling accumulators); K/V fragments
//       reused across both 16-row tiles; P staged via 2KB/wave LDS tile for
//       C-layout -> A-layout conversion.
//   k3: proj = attn @ w_proj^T + bias, 32x64 wave tiles, f32 out.
// Workspace: xb 6.0M | wqb 3.4M | wpb 1.2M | qb/kbf/vtb/ab 4x6.0M  (~34.5 MiB)
// ---------------------------------------------------------------------------

typedef __bf16 bf16_t;
typedef __attribute__((ext_vector_type(16))) __bf16    v16bf;
typedef __attribute__((ext_vector_type(8)))  __bf16    v8bf;
typedef __attribute__((ext_vector_type(8)))  float     v8f;
typedef __attribute__((ext_vector_type(4)))  float     v4f;
typedef __attribute__((ext_vector_type(4)))  unsigned  v4u;

#define S_LEN   4096
#define D_MODEL 768
#define NH      12
#define HD      64
#define QKV_N   2304

static __device__ __forceinline__ unsigned bfbits(float f) {
  unsigned u = __builtin_bit_cast(unsigned, f);
  return u + 0x7FFFu + ((u >> 16) & 1u);  // RNE rounding into bits [31:16]
}

static __device__ __forceinline__ bf16_t f2bf(float f) {
  unsigned short h = (unsigned short)(bfbits(f) >> 16);
  return __builtin_bit_cast(bf16_t, h);
}

static __device__ __forceinline__ unsigned pack2(float lo, float hi) {
  return (bfbits(lo) >> 16) | (bfbits(hi) & 0xFFFF0000u);
}

static __device__ __forceinline__ v8f wmma_bf16(v16bf a, v16bf b, v8f c) {
  // D = A(16x32 bf16) * B(32x16 bf16) + C(16x16 f32)
  return __builtin_amdgcn_wmma_f32_16x16x32_bf16(
      /*neg_a=*/false, a, /*neg_b=*/false, b,
      /*c_mod=*/(short)0, c, /*reuse_a=*/false, /*reuse_b=*/false);
}

// ---- fragment loaders (per ISA 16-bit A 16x32 / B 32x16 VGPR layouts) ------
// A: lane L holds row M=L%16; elems e0..7 -> K = kb + 8*(L/16) + e,
//    elems e8..15 -> K = kb + 16 + 8*(L/16) + (e-8). Two 16B runs per lane.
static __device__ __forceinline__ v16bf load_a_bf16(const bf16_t* base, int row0,
                                                    int stride, int kb, int lane) {
  const bf16_t* p = base + (size_t)(row0 + (lane & 15)) * stride + kb + ((lane >> 4) << 3);
  v8bf lo = *(const v8bf*)(p);
  v8bf hi = *(const v8bf*)(p + 16);
  return __builtin_shufflevector(lo, hi, 0, 1, 2, 3, 4, 5, 6, 7,
                                 8, 9, 10, 11, 12, 13, 14, 15);
}

// B: lane L holds col N=L%16; elems e0..7 -> K = kb + 8*(L/16) + e,
//    elems e8..15 -> K = kb + 16 + 8*(L/16) + (e-8) (same order as A).
// Memory is row-major [N][K] => computes A * B_mem^T (x*W^T, q*k^T, P*vT^T).
static __device__ __forceinline__ v16bf load_b_bf16(const bf16_t* base, int n0,
                                                    int stride, int kb, int lane) {
  const bf16_t* p = base + (size_t)(n0 + (lane & 15)) * stride + kb + ((lane >> 4) << 3);
  v8bf lo = *(const v8bf*)(p);
  v8bf hi = *(const v8bf*)(p + 16);
  return __builtin_shufflevector(lo, hi, 0, 1, 2, 3, 4, 5, 6, 7,
                                 8, 9, 10, 11, 12, 13, 14, 15);
}

// ---------------------------------------------------------------------------
// Kernel 0: f32 -> bf16, 8 elements/thread, b128 in / b128 out.
// ---------------------------------------------------------------------------
__global__ __launch_bounds__(256) void cvt_bf16_kernel(
    const float* __restrict__ src, unsigned* __restrict__ dst, int n) {
  const int i = (blockIdx.x * 256 + threadIdx.x) << 3;
  if (i >= n) return;
  v4f a = *(const v4f*)(src + i);
  v4f b = *(const v4f*)(src + i + 4);
  v4u o;
  o[0] = pack2(a[0], a[1]);
  o[1] = pack2(a[2], a[3]);
  o[2] = pack2(b[0], b[1]);
  o[3] = pack2(b[2], b[3]);
  *(v4u*)(dst + (i >> 1)) = o;
}

// ---------------------------------------------------------------------------
// Kernel 1: qkv = x @ w_qkv^T ; 32x64 tile/wave (8 acc), scatter q/k/vT.
// Tiles: (4096/32)*(2304/64) = 128*36 = 4608 -> 576 blocks of 8 waves.
// N-range of a tile is 64-aligned => (q|k|v, head) uniform per tile.
// ---------------------------------------------------------------------------
__global__ __launch_bounds__(256) void qkv_gemm_kernel(
    const bf16_t* __restrict__ xb, const bf16_t* __restrict__ wqb,
    bf16_t* __restrict__ qb, bf16_t* __restrict__ kbf,
    bf16_t* __restrict__ vtb) {
  const int lane = threadIdx.x & 31;
  const int wave = threadIdx.x >> 5;
  const int tile = blockIdx.x * 8 + wave;
  const int mt = tile / (QKV_N / 64);   // 0..127
  const int nt = tile % (QKV_N / 64);   // 0..35

  v8f acc[2][4] = {};
#pragma unroll 2
  for (int k = 0; k < D_MODEL; k += 32) {
    if (k + 64 < D_MODEL) {  // CDNA5 global_prefetch_b8, two chunks ahead
      __builtin_prefetch(xb + (size_t)(mt * 32 + (lane & 15)) * D_MODEL + k + 64, 0, 1);
      __builtin_prefetch(wqb + (size_t)(nt * 64 + (lane & 15)) * D_MODEL + k + 64, 0, 1);
    }
    v16bf a0 = load_a_bf16(xb, mt * 32,      D_MODEL, k, lane);
    v16bf a1 = load_a_bf16(xb, mt * 32 + 16, D_MODEL, k, lane);
#pragma unroll
    for (int ni = 0; ni < 4; ++ni) {
      v16bf b = load_b_bf16(wqb, nt * 64 + ni * 16, D_MODEL, k, lane);
      acc[0][ni] = wmma_bf16(a0, b, acc[0][ni]);
      acc[1][ni] = wmma_bf16(a1, b, acc[1][ni]);
    }
  }

  // C/D layout: lane L, VGPR r -> M = r + 8*(L/16), N = L%16
  const int n0 = nt * 64;
  const int t = n0 / D_MODEL;           // 0=q, 1=k, 2=v  (uniform)
  const int h = (n0 % D_MODEL) >> 6;    // head           (uniform)
  const int col16 = lane & 15;
  const int row0h = (lane >> 4) << 3;

#pragma unroll
  for (int rt = 0; rt < 2; ++rt) {
    const int srow0 = mt * 32 + rt * 16 + row0h;
#pragma unroll
    for (int ni = 0; ni < 4; ++ni) {
      const int d = ni * 16 + col16;
      if (t == 0) {
        bf16_t* dst = qb + (size_t)h * S_LEN * HD + d;
#pragma unroll
        for (int i = 0; i < 8; ++i)
          dst[(size_t)(srow0 + i) * HD] = f2bf(acc[rt][ni][i] * 0.125f);  // hd^-0.5
      } else if (t == 1) {
        bf16_t* dst = kbf + (size_t)h * S_LEN * HD + d;
#pragma unroll
        for (int i = 0; i < 8; ++i)
          dst[(size_t)(srow0 + i) * HD] = f2bf(acc[rt][ni][i]);
      } else {
        bf16_t* dst = vtb + ((size_t)h * HD + d) * S_LEN;  // transposed
#pragma unroll
        for (int i = 0; i < 8; ++i)
          dst[srow0 + i] = f2bf(acc[rt][ni][i]);
      }
    }
  }
}

// ---------------------------------------------------------------------------
// Kernel 2: fused attention. One wave per (head, 32 query rows).
// Pass 1: row max of q@k^T (recomputed; K B-frags reused across row tiles).
// Pass 2: exp, row sum, out += P@V; V B-frags reused across row tiles; P
// staged through a 32x32 bf16 LDS tile (C-layout -> A-layout).
// 12 heads * 128 qtiles = 1536 waves -> 192 blocks of 8 waves.
// ---------------------------------------------------------------------------
__global__ __launch_bounds__(256) void attn_kernel(
    const bf16_t* __restrict__ qb, const bf16_t* __restrict__ kbf,
    const bf16_t* __restrict__ vtb, bf16_t* __restrict__ ab) {
  __shared__ __align__(16) unsigned short pst[8][32 * 32];  // per-wave P tile

  const int lane = threadIdx.x & 31;
  const int wave = threadIdx.x >> 5;
  const int wid = blockIdx.x * 8 + wave;
  const int head = wid / (S_LEN / 32);
  const int qt = wid % (S_LEN / 32);

  const bf16_t* qh = qb + (size_t)head * S_LEN * HD;
  const bf16_t* kh = kbf + (size_t)head * S_LEN * HD;
  const bf16_t* vh = vtb + (size_t)head * HD * S_LEN;

  // q fragments: 2 row tiles x 2 K-halves (hd=64), held for both passes
  v16bf aq[2][2];
#pragma unroll
  for (int rt = 0; rt < 2; ++rt) {
    aq[rt][0] = load_a_bf16(qh, qt * 32 + rt * 16, HD, 0, lane);
    aq[rt][1] = load_a_bf16(qh, qt * 32 + rt * 16, HD, 32, lane);
  }

  const int col16 = lane & 15;
  const int row0h = (lane >> 4) << 3;

  // ---- pass 1: running row max (lane-local, reduced once at the end) ----
  v8f mrow[2];
#pragma unroll
  for (int rt = 0; rt < 2; ++rt)
#pragma unroll
    for (int i = 0; i < 8; ++i) mrow[rt][i] = -3.0e38f;

  for (int kt = 0; kt < S_LEN / 16; ++kt) {
    v16bf b0 = load_b_bf16(kh, kt * 16, HD, 0, lane);
    v16bf b1 = load_b_bf16(kh, kt * 16, HD, 32, lane);
#pragma unroll
    for (int rt = 0; rt < 2; ++rt) {
      v8f s = {};
      s = wmma_bf16(aq[rt][0], b0, s);
      s = wmma_bf16(aq[rt][1], b1, s);
#pragma unroll
      for (int i = 0; i < 8; ++i) mrow[rt][i] = fmaxf(mrow[rt][i], s[i]);
    }
  }
  // a C-tile row spans 16 contiguous lanes: reduce with xor masks 1..8
#pragma unroll
  for (int rt = 0; rt < 2; ++rt)
#pragma unroll
    for (int i = 0; i < 8; ++i) {
      float v = mrow[rt][i];
      for (int msk = 1; msk <= 8; msk <<= 1) v = fmaxf(v, __shfl_xor(v, msk, 32));
      mrow[rt][i] = v;
    }

  // ---- pass 2: P = exp(s - m), lsum += P, out += P @ V ----
  v8f lsum[2] = {};
  v8f o[2][4] = {};
  unsigned short* myp = &pst[wave][0];

  for (int kc = 0; kc < S_LEN / 32; ++kc) {
#pragma unroll
    for (int sub = 0; sub < 2; ++sub) {
      const int kt = kc * 2 + sub;
      v16bf b0 = load_b_bf16(kh, kt * 16, HD, 0, lane);
      v16bf b1 = load_b_bf16(kh, kt * 16, HD, 32, lane);
      const int col = sub * 16 + col16;
#pragma unroll
      for (int rt = 0; rt < 2; ++rt) {
        v8f s = {};
        s = wmma_bf16(aq[rt][0], b0, s);
        s = wmma_bf16(aq[rt][1], b1, s);
#pragma unroll
        for (int i = 0; i < 8; ++i) {
          float p = __expf(s[i] - mrow[rt][i]);
          lsum[rt][i] += p;
          myp[(rt * 16 + row0h + i) * 32 + col] =
              __builtin_bit_cast(unsigned short, f2bf(p));
        }
      }
    }
    __builtin_amdgcn_wave_barrier();  // order LDS writes before A-frag reads

    v16bf ap0 = load_a_bf16((const bf16_t*)myp, 0, 32, 0, lane);
    v16bf ap1 = load_a_bf16((const bf16_t*)myp, 16, 32, 0, lane);
    const int kbase = kc * 32;
#pragma unroll
    for (int ni = 0; ni < 4; ++ni) {
      v16bf bv = load_b_bf16(vh, ni * 16, S_LEN, kbase, lane);
      o[0][ni] = wmma_bf16(ap0, bv, o[0][ni]);
      o[1][ni] = wmma_bf16(ap1, bv, o[1][ni]);
    }
  }

  // normalize and store attn[s][head*64 + d] (bf16, proj A-operand layout)
#pragma unroll
  for (int rt = 0; rt < 2; ++rt) {
#pragma unroll
    for (int i = 0; i < 8; ++i) {
      float v = lsum[rt][i];
      for (int msk = 1; msk <= 8; msk <<= 1) v += __shfl_xor(v, msk, 32);
      lsum[rt][i] = 1.0f / v;
    }
    const int srow0 = qt * 32 + rt * 16 + row0h;
    bf16_t* dst = ab + (size_t)srow0 * D_MODEL + head * HD + col16;
#pragma unroll
    for (int i = 0; i < 8; ++i) {
      const size_t r = (size_t)i * D_MODEL;
      dst[r + 0]  = f2bf(o[rt][0][i] * lsum[rt][i]);
      dst[r + 16] = f2bf(o[rt][1][i] * lsum[rt][i]);
      dst[r + 32] = f2bf(o[rt][2][i] * lsum[rt][i]);
      dst[r + 48] = f2bf(o[rt][3][i] * lsum[rt][i]);
    }
  }
}

// ---------------------------------------------------------------------------
// Kernel 3: out = attn @ w_proj^T + b ; 32x64 tile/wave, f32 output.
// Tiles: 128 * 12 = 1536 -> 192 blocks of 8 waves.
// ---------------------------------------------------------------------------
__global__ __launch_bounds__(256) void proj_gemm_kernel(
    const bf16_t* __restrict__ ab, const bf16_t* __restrict__ wpb,
    const float* __restrict__ bias, float* __restrict__ out) {
  const int lane = threadIdx.x & 31;
  const int wave = threadIdx.x >> 5;
  const int tile = blockIdx.x * 8 + wave;
  const int mt = tile / (D_MODEL / 64);  // 0..127
  const int nt = tile % (D_MODEL / 64);  // 0..11

  v8f acc[2][4] = {};
#pragma unroll 2
  for (int k = 0; k < D_MODEL; k += 32) {
    if (k + 64 < D_MODEL) {
      __builtin_prefetch(ab + (size_t)(mt * 32 + (lane & 15)) * D_MODEL + k + 64, 0, 1);
      __builtin_prefetch(wpb + (size_t)(nt * 64 + (lane & 15)) * D_MODEL + k + 64, 0, 1);
    }
    v16bf a0 = load_a_bf16(ab, mt * 32,      D_MODEL, k, lane);
    v16bf a1 = load_a_bf16(ab, mt * 32 + 16, D_MODEL, k, lane);
#pragma unroll
    for (int ni = 0; ni < 4; ++ni) {
      v16bf b = load_b_bf16(wpb, nt * 64 + ni * 16, D_MODEL, k, lane);
      acc[0][ni] = wmma_bf16(a0, b, acc[0][ni]);
      acc[1][ni] = wmma_bf16(a1, b, acc[1][ni]);
    }
  }

  const int col16 = lane & 15;
  const int row0h = (lane >> 4) << 3;
#pragma unroll
  for (int rt = 0; rt < 2; ++rt) {
    const int srow0 = mt * 32 + rt * 16 + row0h;
#pragma unroll
    for (int ni = 0; ni < 4; ++ni) {
      const int n = nt * 64 + ni * 16 + col16;
      const float bv = bias[n];
      float* dst = out + (size_t)srow0 * D_MODEL + n;
#pragma unroll
      for (int i = 0; i < 8; ++i) dst[(size_t)i * D_MODEL] = acc[rt][ni][i] + bv;
    }
  }
}

// ---------------------------------------------------------------------------
extern "C" void kernel_launch(void* const* d_in, const int* in_sizes, int n_in,
                              void* d_out, int out_size, void* d_ws, size_t ws_size,
                              hipStream_t stream) {
  const float* x      = (const float*)d_in[0];  // [1,64,64,768]
  const float* w_qkv  = (const float*)d_in[1];  // [2304,768]
  const float* w_proj = (const float*)d_in[2];  // [768,768]
  const float* b_proj = (const float*)d_in[3];  // [768]
  float* out = (float*)d_out;

  char* ws = (char*)d_ws;
  const size_t NX  = (size_t)S_LEN * D_MODEL;       // 3,145,728
  const size_t NWQ = (size_t)QKV_N * D_MODEL;       // 1,769,472
  const size_t NWP = (size_t)D_MODEL * D_MODEL;     //   589,824
  const size_t SEG = (size_t)NH * S_LEN * HD * 2;   // 6 MiB

  bf16_t* xb  = (bf16_t*)(ws);
  bf16_t* wqb = (bf16_t*)(ws + 2 * NX);
  bf16_t* wpb = (bf16_t*)(ws + 2 * (NX + NWQ));
  char*   seg = ws + 2 * (NX + NWQ + NWP);
  bf16_t* qb  = (bf16_t*)(seg + 0 * SEG);
  bf16_t* kbf = (bf16_t*)(seg + 1 * SEG);
  bf16_t* vtb = (bf16_t*)(seg + 2 * SEG);
  bf16_t* ab  = (bf16_t*)(seg + 3 * SEG);

  // one-shot conversions (8 elems/thread)
  cvt_bf16_kernel<<<(int)(NX  / 8 / 256), 256, 0, stream>>>(x,      (unsigned*)xb,  (int)NX);
  cvt_bf16_kernel<<<(int)(NWQ / 8 / 256), 256, 0, stream>>>(w_qkv,  (unsigned*)wqb, (int)NWQ);
  cvt_bf16_kernel<<<(int)(NWP / 8 / 256), 256, 0, stream>>>(w_proj, (unsigned*)wpb, (int)NWP);

  // (S/32)*(QKV_N/64) / 8 waves = 576 blocks
  qkv_gemm_kernel<<<576, 256, 0, stream>>>(xb, wqb, qb, kbf, vtb);
  // NH * (S/32) / 8 = 192 blocks
  attn_kernel<<<192, 256, 0, stream>>>(qb, kbf, vtb, ab);
  // (S/32)*(D/64) / 8 = 192 blocks
  proj_gemm_kernel<<<192, 256, 0, stream>>>(ab, wpb, b_proj, out);
}
